// MaskedCrossAttention_86182813762476
// MI455X (gfx1250) — matrix operand order repaired and checked by reading, not compiled, not run on hardware
//
#include <hip/hip_runtime.h>
#include <hip/hip_bf16.h>
#include <hip/hip_fp16.h>

// MaskedCrossAttention, MI455X (gfx1250).
//
// Structural insight: the segment mask selects at most ONE memory slot per
// token, so softmax is exactly one-hot and the computation reduces to
//   out[b,t,:] = routed ? (memory[b,slot,:] @ Wkv_v) @ Wout : 0
// x / gamma / beta / Wq / sim are mathematically dead. Pipeline:
//   V[1024,512]  = mem_h @ Wkv_v      (WMMA f16)
//   U[1024,2048] = V @ Wout           (WMMA f16)
//   out[b,t,:]   = gather(U, ffill(slot_ids))   (134 MB stream -> HBM bound)

#define B_     4
#define T_     4096
#define S_     256
#define DIM_   2048
#define INNER_ 512

typedef __attribute__((ext_vector_type(16))) _Float16 v16h;
typedef __attribute__((ext_vector_type(8)))  _Float16 v8h;
typedef __attribute__((ext_vector_type(8)))  float    v8f;

// ---------------- f32 -> f16 conversion (row-major copy) ----------------
__global__ void cvt_f32_to_f16_kernel(const float* __restrict__ src,
                                      _Float16* __restrict__ dst, int n) {
  int i = blockIdx.x * blockDim.x + threadIdx.x;
  if (i < n) dst[i] = (_Float16)src[i];
}

// ---------------- pack f32 B[K,N] into WMMA-fragment order --------------
// dst layout: [kt][nt][lane][p] with k = kt*32 + (lane>>4)*16 + p,
//             n = nt*16 + (lane&15).  32 bytes (2x b128) per lane per K-step in GEMM.
// src element = src[k * src_ld + src_col0 + n]  (handles the Wkv V-half).
__global__ void pack_b_kernel(const float* __restrict__ src,
                              _Float16* __restrict__ dst,
                              int K, int N, int src_ld, int src_col0) {
  int i = blockIdx.x * blockDim.x + threadIdx.x;   // over K*N
  if (i >= K * N) return;
  int p    = i & 15;
  int lane = (i >> 4) & 31;
  int tile = i >> 9;                  // kt * (N/16) + nt
  int TN   = N >> 4;
  int kt = tile / TN, nt = tile - kt * TN;
  int k = kt * 32 + (lane >> 4) * 16 + p;
  int n = nt * 16 + (lane & 15);
  dst[i] = (_Float16)src[(size_t)k * src_ld + src_col0 + n];
}

// ---------------- forward-fill slot ids (segmented last-nonzero scan) ----
__global__ void ffill_kernel(const int* __restrict__ ids, int* __restrict__ g) {
  __shared__ int carry[256];
  const int b = blockIdx.x, tid = threadIdx.x;
  const int* row = ids + b * T_;
  int* grow = g + b * T_;
  const int base = tid * 16;

  int vals[16];
  int local = 0;
  #pragma unroll
  for (int j = 0; j < 16; ++j) {
    int v = row[base + j];
    vals[j] = v;
    if (v != 0) local = v;
  }
  carry[tid] = local;
  __syncthreads();

  for (int off = 1; off < 256; off <<= 1) {   // Hillis-Steele, op: b?b:a
    int v = carry[tid];
    int p = (tid >= off) ? carry[tid - off] : 0;
    __syncthreads();
    carry[tid] = (v != 0) ? v : p;
    __syncthreads();
  }
  int run = (tid > 0) ? carry[tid - 1] : 0;

  #pragma unroll
  for (int j = 0; j < 16; ++j) {
    if (vals[j] != 0) run = vals[j];
    grow[base + j] = run - 1;                 // -1 => no route
  }
}

// ---------------- f16 WMMA GEMM: C[M,N] = A[M,K] * Bpacked ----------------
// A row-major f16; B pre-packed by pack_b_kernel. One 16x16 tile per wave.
// Inner loop: 2x global_load_b128 (A) + 2x global_load_b128 (B) + 1 WMMA.
__global__ void wmma_gemm_kernel(const _Float16* __restrict__ A,
                                 const _Float16* __restrict__ Bp,
                                 float* __restrict__ Cf,
                                 _Float16* __restrict__ Ch,
                                 int M, int N, int K, int store_half) {
  const int wid  = (blockIdx.x * blockDim.x + threadIdx.x) >> 5;
  const int lane = threadIdx.x & 31;
  const int tiles_n = N >> 4;
  const int tm = wid / tiles_n;
  const int tn = wid - tm * tiles_n;
  const int mlo = lane & 15;
  const int h   = lane >> 4;

  v8f c = {};
  // Lane's A halves: K in [k0+16h, k0+16h+8) and [k0+16h+8, k0+16h+16).
  const _Float16* __restrict__ arow = A + (size_t)(tm * 16 + mlo) * K + h * 16;
  const _Float16* __restrict__ bptr = Bp + ((size_t)tn * 32 + lane) * 16;
  const size_t bstep = (size_t)tiles_n * 512;        // one kt of packed B

  for (int k0 = 0; k0 < K; k0 += 32) {
    __builtin_prefetch(arow + k0 + 32, 0, 0);        // -> global_prefetch_b8
    v8h lo = *(const v8h*)(arow + k0);
    v8h hi = *(const v8h*)(arow + k0 + 8);
    v16h a = __builtin_shufflevector(lo, hi,
        0, 1, 2, 3, 4, 5, 6, 7, 8, 9, 10, 11, 12, 13, 14, 15);
    v16h bf = *(const v16h*)bptr;
    bptr += bstep;
    c = __builtin_amdgcn_wmma_f32_16x16x32_f16(
        /*neg_a=*/false, a, /*neg_b=*/false, bf,
        /*c_mod=*/(short)0, c, /*reuse_a=*/false, /*reuse_b=*/false);
  }

  // C/D layout: VGPR r holds row M = 8h + r, column N = lane&15.
  const int row0 = tm * 16 + h * 8;
  const int col  = tn * 16 + mlo;
  if (store_half) {
    #pragma unroll
    for (int r = 0; r < 8; ++r)
      Ch[(size_t)(row0 + r) * N + col] = (_Float16)c[r];
  } else {
    #pragma unroll
    for (int r = 0; r < 8; ++r)
      Cf[(size_t)(row0 + r) * N + col] = c[r];
  }
}

// ---------------- one-hot gather: out[b,t,:] = U[b,slot,:] or 0 ----------
__global__ void gather_out_kernel(const float* __restrict__ U,
                                  const int* __restrict__ g,
                                  const unsigned char* __restrict__ mmask,
                                  float* __restrict__ out) {
  const int bt = blockIdx.x;                 // 0 .. B*T-1
  const int b  = bt >> 12;                   // T = 4096
  const int slot = g[bt];
  const bool ok = (slot >= 0) && (mmask[b * S_ + slot] != 0);
  const float4* __restrict__ src =
      (const float4*)(U + ((size_t)b * S_ + (slot < 0 ? 0 : slot)) * DIM_);
  float4* __restrict__ dst = (float4*)(out + (size_t)bt * DIM_);
  const float4 z = make_float4(0.f, 0.f, 0.f, 0.f);
  const int t = threadIdx.x;                 // 256 threads, DIM/4 = 512 float4
  #pragma unroll
  for (int i = 0; i < 2; ++i) {
    int idx = t + i * 256;
    dst[idx] = ok ? src[idx] : z;
  }
}

extern "C" void kernel_launch(void* const* d_in, const int* in_sizes, int n_in,
                              void* d_out, int out_size, void* d_ws, size_t ws_size,
                              hipStream_t stream) {
  // setup_inputs order: x, memory, placeholder_slot_ids, memory_mask,
  //                     gamma, beta, Wq, Wkv, Wout
  const float* memory = (const float*)d_in[1];
  const int*   sids   = (const int*)d_in[2];
  const unsigned char* mmask = (const unsigned char*)d_in[3];   // jnp bool -> u8
  const float* Wkv    = (const float*)d_in[7];
  const float* Wout   = (const float*)d_in[8];
  float* out = (float*)d_out;
  (void)in_sizes; (void)n_in; (void)out_size; (void)ws_size;

  // Workspace layout (~17.1 MB used)
  char* ws = (char*)d_ws;
  _Float16* mem_h  = (_Float16*)(ws);                 // 4 MB  [1024, 2048] f16
  _Float16* wv_p   = (_Float16*)(ws + (4u  << 20));   // 2 MB  packed B (2048x512)
  _Float16* wout_p = (_Float16*)(ws + (6u  << 20));   // 2 MB  packed B (512x2048)
  _Float16* v_h    = (_Float16*)(ws + (8u  << 20));   // 1 MB  [1024, 512] f16
  float*    U      = (float*)   (ws + (9u  << 20));   // 8 MB  [1024, 2048] f32
  int*      g      = (int*)     (ws + (17u << 20));   // 64 KB [B, T] int

  const int nMem = B_ * S_ * DIM_;       // 2,097,152
  const int nWv  = DIM_ * INNER_;        // 1,048,576
  const int nWo  = INNER_ * DIM_;        // 1,048,576

  cvt_f32_to_f16_kernel<<<nMem / 256, 256, 0, stream>>>(memory, mem_h, nMem);
  // Wkv V-half: K=2048, N=512, row stride 1024, col offset 512.
  pack_b_kernel<<<nWv / 256, 256, 0, stream>>>(Wkv,  wv_p,   DIM_,   INNER_, 1024, 512);
  // Wout: K=512, N=2048, row stride 2048.
  pack_b_kernel<<<nWo / 256, 256, 0, stream>>>(Wout, wout_p, INNER_, DIM_,   DIM_, 0);

  ffill_kernel<<<B_, 256, 0, stream>>>(sids, g);

  // V[1024,512] = mem_h[1024,2048] @ Wv   (store f16)
  {
    int M = B_ * S_, N = INNER_, K = DIM_;
    int waves = (M / 16) * (N / 16);                 // 2048 -> 256 blocks
    wmma_gemm_kernel<<<waves * 32 / 256, 256, 0, stream>>>(
        mem_h, wv_p, nullptr, v_h, M, N, K, 1);
  }
  // U[1024,2048] = v_h[1024,512] @ Wout   (store f32)
  {
    int M = B_ * S_, N = DIM_, K = INNER_;
    int waves = (M / 16) * (N / 16);                 // 8192 -> 1024 blocks
    wmma_gemm_kernel<<<waves * 32 / 256, 256, 0, stream>>>(
        v_h, wout_p, U, nullptr, M, N, K, 0);
  }

  gather_out_kernel<<<B_ * T_, 256, 0, stream>>>(U, g, mmask, out);
}
